// SelfAttention_30605936951804
// MI455X (gfx1250) — matrix operand order repaired and checked by reading, not compiled, not run on hardware
//
#include <hip/hip_runtime.h>
#include <hip/hip_bf16.h>

// ---------------------------------------------------------------------------
// Self-attention (N=4, S=2048, E=1024, H=16, D=64) for gfx1250 (MI455X).
// All matmuls via v_wmma_f32_16x16x32_bf16 (wave32, 16x16 C tiles, f32 acc).
// ---------------------------------------------------------------------------

#define NB   4
#define SEQ  2048
#define HEADS 16
#define HD   64
#define EMB  1024

typedef __attribute__((ext_vector_type(16))) __bf16 v16bf;
typedef __attribute__((ext_vector_type(8)))  __bf16 v8bf;
typedef __attribute__((ext_vector_type(8)))  float  v8f;

__device__ __forceinline__ v8f wmma_bf16(v16bf a, v16bf b, v8f c) {
  return __builtin_amdgcn_wmma_f32_16x16x32_bf16(
      /*neg_a=*/false, a, /*neg_b=*/false, b,
      /*c_mod=*/(short)0, c, /*reuse_a=*/false, /*reuse_b=*/false);
}

// A-matrix (16 rows x 32 K, bf16) from a row-major [rows x ld] bf16 buffer.
// Contiguous K order: lanes 0-15 hold row (lane), K = 0..15;
//                     lanes 16-31 hold row (lane-16), K = 16..31.
__device__ __forceinline__ v16bf load_A_bf16(const __bf16* base, int ld,
                                             int row0, int k0, int lane) {
  const int r  = lane & 15;
  const int hi = (lane >> 4) & 1;
  const __bf16* p = base + (size_t)(row0 + r) * ld + k0 + hi * 16;
  v8bf c0 = *(const v8bf*)p;        // K = k0 + hi*16 .. +7
  v8bf c1 = *(const v8bf*)(p + 8);  // K = k0 + hi*16 + 8 .. +15
  v16bf a;
#pragma unroll
  for (int i = 0; i < 8; ++i) { a[i] = c0[i]; a[i + 8] = c1[i]; }
  return a;
}

// B-matrix (32 K x 16 cols, bf16) from a row-major [cols x ld] buffer where
// each B-column's K-run is a contiguous row (i.e. buffer[n][k] = B[k][n]).
// Lanes 0-15: col n=lane, K = k0..k0+15; lanes 16-31: K = k0+16..k0+31.
__device__ __forceinline__ v16bf load_B_bf16(const __bf16* base, int ld,
                                             int n0, int k0, int lane) {
  const int n  = lane & 15;
  const int hi = (lane >> 4) & 1;
  const __bf16* p = base + (size_t)(n0 + n) * ld + k0 + hi * 16;
  v8bf c0 = *(const v8bf*)p;
  v8bf c1 = *(const v8bf*)(p + 8);
  v16bf b;
#pragma unroll
  for (int i = 0; i < 8; ++i) { b[i] = c0[i]; b[i + 8] = c1[i]; }
  return b;
}

// Same loaders from f32 sources (convert to bf16 in registers).
__device__ __forceinline__ v16bf load_A_f32(const float* base, int ld,
                                            int row0, int k0, int lane) {
  const int r  = lane & 15;
  const int hi = (lane >> 4) & 1;
  const float* p = base + (size_t)(row0 + r) * ld + k0 + hi * 16;
  v16bf a;
#pragma unroll
  for (int i = 0; i < 8; ++i) {
    a[i]     = (__bf16)p[i];
    a[i + 8] = (__bf16)p[i + 8];
  }
  return a;
}

__device__ __forceinline__ v16bf load_B_f32(const float* base, int ld,
                                            int n0, int k0, int lane) {
  const int n  = lane & 15;
  const int hi = (lane >> 4) & 1;
  const float* p = base + (size_t)(n0 + n) * ld + k0 + hi * 16;
  v16bf b;
#pragma unroll
  for (int i = 0; i < 16; ++i) b[i] = (__bf16)p[i];
  return b;
}

// ---------------------------------------------------------------------------
// Kernel 1: per-head projections. X:(N*S*H, 64) f32, W:(64,64) f32,
// out = bf16(X @ W^T). Q,K -> (N,H,S,D); V -> transposed (N,H,D,S).
// One wave = 16 rows (the 16 heads of one (n,s)) x all 64 output cols.
// ---------------------------------------------------------------------------
__global__ void proj_kernel(const float* __restrict__ q,
                            const float* __restrict__ kk,
                            const float* __restrict__ vv,
                            const float* __restrict__ Wq,
                            const float* __restrict__ Wk,
                            const float* __restrict__ Wv,
                            __bf16* __restrict__ qp,
                            __bf16* __restrict__ kp,
                            __bf16* __restrict__ vpT) {
  const int lane = threadIdx.x & 31;
  const int wave = threadIdx.x >> 5;
  const int gw   = blockIdx.x * 8 + wave;       // 0 .. 3*8192-1
  const int which = gw / (NB * SEQ);            // 0=Q 1=K 2=V
  const int rt    = gw % (NB * SEQ);            // flat (n*S + s)
  const int n = rt / SEQ;
  const int s = rt % SEQ;
  const int g0 = rt * HEADS;                    // 16 consecutive rows = heads

  const float* X = (which == 0) ? q : (which == 1) ? kk : vv;
  const float* W = (which == 0) ? Wq : (which == 1) ? Wk : Wv;

  v16bf a0 = load_A_f32(X, HD, g0, 0, lane);
  v16bf a1 = load_A_f32(X, HD, g0, 32, lane);

  const int ln = lane & 15;
  const int hi = (lane >> 4) & 1;

#pragma unroll
  for (int et = 0; et < 4; ++et) {              // output cols e = et*16..+15
    v16bf b0 = load_B_f32(W, HD, et * 16, 0, lane);
    v16bf b1 = load_B_f32(W, HD, et * 16, 32, lane);
    v8f acc = {};
    acc = wmma_bf16(a0, b0, acc);
    acc = wmma_bf16(a1, b1, acc);
    if (which < 2) {
      __bf16* out = (which == 0) ? qp : kp;     // (N,H,S,D)
#pragma unroll
      for (int r = 0; r < 8; ++r) {
        int h = r + 8 * hi;
        out[(((size_t)n * HEADS + h) * SEQ + s) * HD + et * 16 + ln] =
            (__bf16)acc[r];
      }
    } else {                                    // V transposed: (N,H,D,S)
#pragma unroll
      for (int r = 0; r < 8; ++r) {
        int h = r + 8 * hi;
        vpT[(((size_t)n * HEADS + h) * HD + et * 16 + ln) * SEQ + s] =
            (__bf16)acc[r];
      }
    }
  }
}

// ---------------------------------------------------------------------------
// Kernel 2: Wo f32 -> bf16.
// ---------------------------------------------------------------------------
__global__ void cvt_wo_kernel(const float* __restrict__ Wo,
                              __bf16* __restrict__ wo_bf) {
  int i = blockIdx.x * blockDim.x + threadIdx.x;
  if (i < EMB * EMB) wo_bf[i] = (__bf16)Wo[i];
}

// ---------------------------------------------------------------------------
// Kernel 3: flash attention. One wave = (n, h, 16-query tile), streams K/V in
// chunks of 32 key columns with online softmax. Scores/PV via WMMA; P tile is
// transposed C-layout -> A-layout through per-wave LDS.
// ---------------------------------------------------------------------------
__global__ void attn_kernel(const __bf16* __restrict__ qp,
                            const __bf16* __restrict__ kp,
                            const __bf16* __restrict__ vpT,
                            const int* __restrict__ mask,
                            __bf16* __restrict__ xout) {
  __shared__ __align__(16) __bf16 smem[8][16 * 32];

  const int lane = threadIdx.x & 31;
  const int wave = threadIdx.x >> 5;
  const int gw   = blockIdx.x * 8 + wave;       // 0 .. 8191
  const int qt = gw & (SEQ / 16 - 1);           // query tile 0..127
  const int nh = gw >> 7;                       // (n*16 + h), 0..63
  const int q0 = qt * 16;
  const int n  = nh >> 4;
  const int h  = nh & 15;
  const int ln = lane & 15;
  const int hi = (lane >> 4) & 1;

  const __bf16* qbase = qp  + (size_t)nh * SEQ * HD;
  const __bf16* kbase = kp  + (size_t)nh * SEQ * HD;
  const __bf16* vbase = vpT + (size_t)nh * HD * SEQ;
  const int*    mbase = mask + (size_t)n * SEQ * SEQ;

  v16bf qa0 = load_A_bf16(qbase, HD, q0, 0, lane);
  v16bf qa1 = load_A_bf16(qbase, HD, q0, 32, lane);

  float mrun[8], lsum[8];
  v8f o0 = {}, o1 = {}, o2 = {}, o3 = {};
#pragma unroll
  for (int r = 0; r < 8; ++r) { mrun[r] = -3.0e38f; lsum[r] = 0.0f; }

  __bf16* pl = smem[wave];

  for (int k0 = 0; k0 < SEQ; k0 += 32) {
    // ---- scores S = Q @ K^T for 32 key columns (two 16-col tiles) ----
    v8f st0, st1;
    {
      v16bf kb0 = load_B_bf16(kbase, HD, k0, 0, lane);
      v16bf kb1 = load_B_bf16(kbase, HD, k0, 32, lane);
      v8f s = {};
      s = wmma_bf16(qa0, kb0, s);
      s = wmma_bf16(qa1, kb1, s);
      st0 = s;
    }
    {
      v16bf kb0 = load_B_bf16(kbase, HD, k0 + 16, 0, lane);
      v16bf kb1 = load_B_bf16(kbase, HD, k0 + 16, 32, lane);
      v8f s = {};
      s = wmma_bf16(qa0, kb0, s);
      s = wmma_bf16(qa1, kb1, s);
      st1 = s;
    }

    // ---- mask (before scale, as reference) + scale by 1/sqrt(E)=1/32 ----
#pragma unroll
    for (int r = 0; r < 8; ++r) {
      int row = q0 + r + 8 * hi;
      float e0 = st0[r], e1 = st1[r];
      if (mbase[(size_t)row * SEQ + k0 + ln] == 0)      e0 = -1.0e10f;
      if (mbase[(size_t)row * SEQ + k0 + 16 + ln] == 0) e1 = -1.0e10f;
      st0[r] = e0 * 0.03125f;
      st1[r] = e1 * 0.03125f;
    }

    // ---- online softmax: row reductions across 16-lane halves ----
#pragma unroll
    for (int r = 0; r < 8; ++r) {
      float a = fmaxf(st0[r], st1[r]);
      a = fmaxf(a, __shfl_xor(a, 1, 32));
      a = fmaxf(a, __shfl_xor(a, 2, 32));
      a = fmaxf(a, __shfl_xor(a, 4, 32));
      a = fmaxf(a, __shfl_xor(a, 8, 32));
      float mnew = fmaxf(mrun[r], a);
      float corr = __expf(mrun[r] - mnew);
      mrun[r] = mnew;
      float p0 = __expf(st0[r] - mnew);
      float p1 = __expf(st1[r] - mnew);
      st0[r] = p0;
      st1[r] = p1;
      float ps = p0 + p1;
      ps += __shfl_xor(ps, 1, 32);
      ps += __shfl_xor(ps, 2, 32);
      ps += __shfl_xor(ps, 4, 32);
      ps += __shfl_xor(ps, 8, 32);
      lsum[r] = lsum[r] * corr + ps;
      o0[r] *= corr; o1[r] *= corr; o2[r] *= corr; o3[r] *= corr;
    }

    // ---- P: C-layout -> A-layout via LDS (16 x 32 bf16 tile) ----
#pragma unroll
    for (int r = 0; r < 8; ++r) {
      pl[(r + 8 * hi) * 32 + ln]      = (__bf16)st0[r];
      pl[(r + 8 * hi) * 32 + 16 + ln] = (__bf16)st1[r];
    }
    asm volatile("s_wait_dscnt 0" ::: "memory");
    v16bf pa = load_A_bf16(pl, 32, 0, 0, lane);

    // ---- O += P @ V  (V^T rows are contiguous along S) ----
    {
      v16bf vb = load_B_bf16(vbase, SEQ, 0,  k0, lane);
      o0 = wmma_bf16(pa, vb, o0);
    }
    {
      v16bf vb = load_B_bf16(vbase, SEQ, 16, k0, lane);
      o1 = wmma_bf16(pa, vb, o1);
    }
    {
      v16bf vb = load_B_bf16(vbase, SEQ, 32, k0, lane);
      o2 = wmma_bf16(pa, vb, o2);
    }
    {
      v16bf vb = load_B_bf16(vbase, SEQ, 48, k0, lane);
      o3 = wmma_bf16(pa, vb, o3);
    }
  }

  // ---- normalize and store X = softmax(S) @ V as (N,S,E) bf16 ----
#pragma unroll
  for (int r = 0; r < 8; ++r) {
    int row = q0 + r + 8 * hi;
    size_t base = ((size_t)n * SEQ + row) * EMB + h * HD + ln;
    float inv = 1.0f / lsum[r];
    xout[base + 0]  = (__bf16)(o0[r] * inv);
    xout[base + 16] = (__bf16)(o1[r] * inv);
    xout[base + 32] = (__bf16)(o2[r] * inv);
    xout[base + 48] = (__bf16)(o3[r] * inv);
  }
}

// ---------------------------------------------------------------------------
// Kernel 4: out = X @ Wo^T + bo. X:(8192,1024) bf16, Wo bf16 row-major (its
// rows are exactly the B-column K-runs). One wave = one 16x64 output tile:
// each A-tile load feeds 4 WMMAs (4x reuse), 4x less L2 re-read of X.
// ---------------------------------------------------------------------------
__global__ void ogemm_kernel(const __bf16* __restrict__ x,
                             const __bf16* __restrict__ wo,
                             const float* __restrict__ bo,
                             float* __restrict__ out) {
  const int lane = threadIdx.x & 31;
  const int wave = threadIdx.x >> 5;
  const int gw   = blockIdx.x * 8 + wave;   // 0 .. 8191
  const int ng = gw & 15;                   // 16 column groups (64 cols each)
  const int mt = gw >> 4;                   // 512 row tiles
  const int m0 = mt * 16, n0 = ng * 64;

  v8f acc0 = {}, acc1 = {}, acc2 = {}, acc3 = {};
  for (int c0 = 0; c0 < EMB; c0 += 32) {
    if (c0 + 32 < EMB) {
      __builtin_prefetch(x + (size_t)(m0 + (lane & 15)) * EMB + c0 + 32, 0, 1);
    }
    v16bf a = load_A_bf16(x, EMB, m0, c0, lane);
    {
      v16bf b = load_B_bf16(wo, EMB, n0, c0, lane);
      acc0 = wmma_bf16(a, b, acc0);
    }
    {
      v16bf b = load_B_bf16(wo, EMB, n0 + 16, c0, lane);
      acc1 = wmma_bf16(a, b, acc1);
    }
    {
      v16bf b = load_B_bf16(wo, EMB, n0 + 32, c0, lane);
      acc2 = wmma_bf16(a, b, acc2);
    }
    {
      v16bf b = load_B_bf16(wo, EMB, n0 + 48, c0, lane);
      acc3 = wmma_bf16(a, b, acc3);
    }
  }

  const int ln = lane & 15;
  const int hi = (lane >> 4) & 1;
  float b0 = bo[n0 + ln];
  float b1 = bo[n0 + 16 + ln];
  float b2 = bo[n0 + 32 + ln];
  float b3 = bo[n0 + 48 + ln];
#pragma unroll
  for (int r = 0; r < 8; ++r) {
    size_t base = (size_t)(m0 + r + 8 * hi) * EMB + n0 + ln;
    out[base + 0]  = acc0[r] + b0;
    out[base + 16] = acc1[r] + b1;
    out[base + 32] = acc2[r] + b2;
    out[base + 48] = acc3[r] + b3;
  }
}

// ---------------------------------------------------------------------------
extern "C" void kernel_launch(void* const* d_in, const int* in_sizes, int n_in,
                              void* d_out, int out_size, void* d_ws,
                              size_t ws_size, hipStream_t stream) {
  const float* q    = (const float*)d_in[0];
  const float* kk   = (const float*)d_in[1];
  const float* vv   = (const float*)d_in[2];
  const int*   mask = (const int*)d_in[3];
  const float* Wq   = (const float*)d_in[4];
  const float* Wk   = (const float*)d_in[5];
  const float* Wv   = (const float*)d_in[6];
  const float* Wo   = (const float*)d_in[7];
  const float* bo   = (const float*)d_in[8];
  float* out = (float*)d_out;

  char* ws = (char*)d_ws;
  const size_t tensor_bytes = (size_t)NB * SEQ * EMB * sizeof(__bf16); // 16 MB
  __bf16* qp    = (__bf16*)(ws);
  __bf16* kp    = (__bf16*)(ws + tensor_bytes);
  __bf16* vpT   = (__bf16*)(ws + 2 * tensor_bytes);
  __bf16* xattn = (__bf16*)(ws + 3 * tensor_bytes);
  __bf16* wo_bf = (__bf16*)(ws + 4 * tensor_bytes);

  // 1) projections: 3 * (N*S) row-tiles, 8 waves/block
  proj_kernel<<<3 * NB * SEQ / 8, 256, 0, stream>>>(q, kk, vv, Wq, Wk, Wv,
                                                    qp, kp, vpT);
  // 2) Wo -> bf16
  cvt_wo_kernel<<<(EMB * EMB) / 256, 256, 0, stream>>>(Wo, wo_bf);
  // 3) flash attention: N*H*(S/16) = 8192 waves
  attn_kernel<<<(NB * HEADS * (SEQ / 16)) / 8, 256, 0, stream>>>(
      qp, kp, vpT, mask, xattn);
  // 4) output GEMM: 512 row tiles x 16 col groups = 8192 waves
  ogemm_kernel<<<((NB * SEQ / 16) * (EMB / 64)) / 8, 256, 0, stream>>>(
      xattn, wo_bf, bo, out);
}
